// BoltzmannTrafficFlow_87565793230934
// MI455X (gfx1250) — compile-verified, matrix operand order checked by reading, not run on hardware
//
#include <hip/hip_runtime.h>
#include <hip/hip_bf16.h>

typedef __attribute__((ext_vector_type(16))) _Float16 v16h;
typedef __attribute__((ext_vector_type(8)))  float    v8f;

// Problem constants (match reference)
#define NN      10000
#define EE      160000
#define BB      4
#define T_IN    12
#define DD      2
#define HH      128
#define QQ      15
#define OUT_LEN 12
#define DT      0.1f
#define EPSF    1e-6f
#define BN      (BB * NN)       // 40000
#define MTILES  (BN / 16)       // 2500 exact

// Padded leading dims (enable aligned float4 loads, mask-free fragments)
#define LD_F    16              // f / omega / flow rows (col 15 kept zero)
#define LD_C1   8               // cat1 rows (cols 6,7 kept zero)
#define LD_C2   384             // cat2 rows (multiple of 32)

// ---------------------------------------------------------------------------
// Fragment helpers. CDNA5 16-bit A-fragment layout (ISA 7.12.2):
//   lane l holds row (l&15); element i covers
//   k = kbase + (i&7) + ((l&16)?8:0) + ((i&8)?16:0)
// ---------------------------------------------------------------------------
__device__ __forceinline__ void put8(v16h& a, int base, const float4& x,
                                     const float4& y) {
    a[base + 0] = (_Float16)x.x; a[base + 1] = (_Float16)x.y;
    a[base + 2] = (_Float16)x.z; a[base + 3] = (_Float16)x.w;
    a[base + 4] = (_Float16)y.x; a[base + 5] = (_Float16)y.y;
    a[base + 6] = (_Float16)y.z; a[base + 7] = (_Float16)y.w;
}

// Full 32-wide K chunk, everything in-bounds (K multiple of 32 region).
__device__ __forceinline__ v16h load_a_full(const float* rowp, int koff) {
    const float4* p = (const float4*)(rowp + koff);        // k: koff+0..7
    const float4* q = (const float4*)(rowp + koff + 16);   // k: koff+16..23
    float4 a0 = p[0], a1 = p[1], b0 = q[0], b1 = q[1];
    v16h a;
    put8(a, 0, a0, a1);
    put8(a, 8, b0, b1);
    return a;
}

// K <= 16, row padded to >=16 floats with zeros: upper half (k=16..31) == 0.
__device__ __forceinline__ v16h load_a_k16(const float* rowp, int koff) {
    const float4* p = (const float4*)(rowp + koff);
    float4 a0 = p[0], a1 = p[1];
    v16h a;
    put8(a, 0, a0, a1);
#pragma unroll
    for (int i = 8; i < 16; ++i) a[i] = (_Float16)0.0f;
    return a;
}

// K <= 8, row padded to 8 floats: lanes with koff=8 contribute zeros.
__device__ __forceinline__ v16h load_a_k8(const float* rowp, int koff) {
    v16h a;
#pragma unroll
    for (int i = 0; i < 16; ++i) a[i] = (_Float16)0.0f;
    if (koff == 0) {
        const float4* p = (const float4*)rowp;
        float4 a0 = p[0], a1 = p[1];
        put8(a, 0, a0, a1);
    }
    return a;
}

// activation: 0 none, 1 relu, 2 tanh, 3 softplus
template <int ACT>
__device__ __forceinline__ float activate(float x) {
    if (ACT == 1) return x > 0.0f ? x : 0.0f;
    if (ACT == 2) return tanhf(x);
    if (ACT == 3) return (x > 20.0f) ? x : log1pf(expf(x));
    return x;
}

// ---------------------------------------------------------------------------
// Pack B[K,Ncols] (f32, row-major) into f16 fragment-major layout:
// frag index = (kc*NT + jt), then [lane][16] contiguous (32 bytes per lane).
// ---------------------------------------------------------------------------
__global__ void pack_b(const float* __restrict__ Bm, int Ncols, int K,
                       int KT, int NT, _Float16* __restrict__ out) {
    int idx = blockIdx.x * blockDim.x + threadIdx.x;
    if (idx >= KT * NT * 32) return;
    int lane = idx & 31;
    int frag = idx >> 5;
    int jt = frag % NT;
    int kc = frag / NT;
    int n = jt * 16 + (lane & 15);
    _Float16* o = out + (size_t)idx * 16;
#pragma unroll
    for (int i = 0; i < 16; ++i) {
        int k = kc * 32 + (i & 7) + ((lane & 16) ? 8 : 0) + ((i & 8) ? 16 : 0);
        float v = (k < K && n < Ncols) ? Bm[(size_t)k * Ncols + n] : 0.0f;
        o[i] = (_Float16)v;
    }
}

// ---------------------------------------------------------------------------
// Wide GEMM: C[M,128] = act(A[M,K] @ B + bias).  grid = MTILES, block = 256
// (8 waves; wave w owns column tile w).  Bp layout: [KT][8][32] v16h frags.
// AMODE: 0 full chunks, 1 K<=16 padded row, 2 K<=8 padded row.
// LDA/LDC compile-time -> immediate-offset address arithmetic.
// ---------------------------------------------------------------------------
template <int ACT, int K, int AMODE, int LDA, int LDC>
__global__ void gemm_wide(const float* __restrict__ A,
                          const v16h* __restrict__ Bp,
                          const float* __restrict__ bias,
                          float* __restrict__ C) {
    const int lane = threadIdx.x & 31;
    const int w    = threadIdx.x >> 5;
    const int mrow = blockIdx.x * 16;
    const int m    = mrow + (lane & 15);
    const int koff = (lane & 16) ? 8 : 0;
    const float* rowp = A + (size_t)m * LDA;
    v8f acc = {};
    constexpr int KT = (K + 31) / 32;
#pragma unroll
    for (int kc = 0; kc < KT; ++kc) {
        v16h a;
        if (AMODE == 0)      a = load_a_full(rowp + kc * 32, koff);
        else if (AMODE == 1) a = load_a_k16(rowp, koff);
        else                 a = load_a_k8(rowp, koff);
        v16h b = Bp[(kc * 8 + w) * 32 + lane];
        acc = __builtin_amdgcn_wmma_f32_16x16x32_f16(false, a, false, b,
                                                     (short)0, acc, false, false);
    }
    const int n  = w * 16 + (lane & 15);
    const float bn = bias[n];
    float* cp = C + (size_t)(mrow + ((lane & 16) ? 8 : 0)) * LDC + n;
#pragma unroll
    for (int r = 0; r < 8; ++r)
        cp[(size_t)r * LDC] = activate<ACT>(acc[r] + bn);
}

// ---------------------------------------------------------------------------
// Narrow GEMM: C[M,LDC] = act(A[M,K] @ B[K,Ncols] + bias), cols >= Ncols
// zero-filled.  K multiple of 32.  block = 256 (8 waves = 8 row tiles).
// ---------------------------------------------------------------------------
template <int ACT, int K, int LDA, int LDC>
__global__ void gemm_narrow(const float* __restrict__ A,
                            const v16h* __restrict__ Bp,
                            const float* __restrict__ bias, int Ncols,
                            float* __restrict__ C) {
    const int lane  = threadIdx.x & 31;
    const int w     = threadIdx.x >> 5;
    const int mtile = blockIdx.x * 8 + w;
    if (mtile >= MTILES) return;                  // wave-uniform exit
    const int m    = mtile * 16 + (lane & 15);
    const int koff = (lane & 16) ? 8 : 0;
    const float* rowp = A + (size_t)m * LDA;
    v8f acc = {};
    constexpr int KT = K / 32;
#pragma unroll
    for (int kc = 0; kc < KT; ++kc) {
        v16h a = load_a_full(rowp + kc * 32, koff);
        v16h b = Bp[kc * 32 + lane];
        acc = __builtin_amdgcn_wmma_f32_16x16x32_f16(false, a, false, b,
                                                     (short)0, acc, false, false);
    }
    const int n = lane & 15;
    const bool nv = (n < Ncols);
    const float bn = nv ? bias[n] : 0.0f;
    float* cp = C + (size_t)(mtile * 16 + ((lane & 16) ? 8 : 0)) * LDC + n;
#pragma unroll
    for (int r = 0; r < 8; ++r)
        cp[(size_t)r * LDC] = nv ? activate<ACT>(acc[r] + bn) : 0.0f;
}

// ---------------------------------------------------------------------------
// Graph / elementwise kernels
// ---------------------------------------------------------------------------
__global__ void deg_wsum_kernel(const int* __restrict__ dst,
                                const float* __restrict__ ew,
                                float* __restrict__ deg, float* __restrict__ wsum) {
    int e = blockIdx.x * blockDim.x + threadIdx.x;
    if (e < EE) {
        int d = dst[e];
        atomicAdd(&deg[d], 1.0f);
        atomicAdd(&wsum[d], ew[e]);
    }
}

__global__ void norm_kernel(const float* __restrict__ deg, float* __restrict__ norm) {
    int n = blockIdx.x * blockDim.x + threadIdx.x;
    if (n < NN) norm[n] = rsqrtf(fmaxf(deg[n], 1.0f));
}

__global__ void alpha_kernel(const float* __restrict__ ew, const int* __restrict__ dst,
                             const float* __restrict__ wsum, float* __restrict__ alpha) {
    int e = blockIdx.x * blockDim.x + threadIdx.x;
    if (e < EE) alpha[e] = ew[e] / fmaxf(wsum[dst[e]], EPSF);
}

// cat1[b*N+n, 0:2] = inputs[b, T_IN-1, n, :]; cols 2..7 = 0
__global__ void extract_kernel(const float* __restrict__ inputs,
                               float* __restrict__ cat1) {
    int idx = blockIdx.x * blockDim.x + threadIdx.x;  // over B*N
    if (idx < BN) {
        int b = idx / NN, n = idx - b * NN;
        size_t s = ((size_t)(b * T_IN + (T_IN - 1)) * NN + n) * DD;
        float* row = cat1 + (size_t)idx * LD_C1;
        row[0] = inputs[s + 0];
        row[1] = inputs[s + 1];
        row[2] = 0.0f; row[3] = 0.0f; row[4] = 0.0f;
        row[5] = 0.0f; row[6] = 0.0f; row[7] = 0.0f;
    }
}

// buf[(b,dst),dstOff+c] += norm[src] * buf[(b,src),srcOff+c]
// grid: (ceil(E*F/256), B);  F power of two -> shift/mask indexing.
template <int F, int LD>
__global__ void hop_scatter(float* __restrict__ buf, int srcOff, int dstOff,
                            const float* __restrict__ norm,
                            const int* __restrict__ src, const int* __restrict__ dst) {
    int idx = blockIdx.x * blockDim.x + threadIdx.x;
    if (idx >= EE * F) return;
    int b = blockIdx.y;
    int e = idx >> __builtin_ctz(F);
    int c = idx & (F - 1);
    int s = src[e], d = dst[e];
    float v = norm[s] * buf[((size_t)b * NN + s) * LD + srcOff + c];
    atomicAdd(&buf[((size_t)b * NN + d) * LD + dstOff + c], v);
}

// buf[(b,n),off+c] *= norm[n]     grid: (ceil(N*F/256), B)
template <int F, int LD>
__global__ void hop_scale(float* __restrict__ buf, int off,
                          const float* __restrict__ norm) {
    int idx = blockIdx.x * blockDim.x + threadIdx.x;
    if (idx >= NN * F) return;
    int b = blockIdx.y;
    int n = idx >> __builtin_ctz(F);
    int c = idx & (F - 1);
    buf[((size_t)b * NN + n) * LD + off + c] *= norm[n];
}

// flow[(b,dst),q] += alpha[e] * f[(b,src),q]   grid: (ceil(E/256), B)
__global__ void flow_scatter(const float* __restrict__ f, const float* __restrict__ alpha,
                             const int* __restrict__ src, const int* __restrict__ dst,
                             float* __restrict__ flow) {
    int e = blockIdx.x * blockDim.x + threadIdx.x;
    if (e >= EE) return;
    int b = blockIdx.y;
    int s = src[e], d = dst[e];
    float a = alpha[e];
    const float4* fs = (const float4*)(f + ((size_t)b * NN + s) * LD_F);
    float4 r0 = fs[0], r1 = fs[1], r2 = fs[2], r3 = fs[3];
    float fv[16] = {r0.x, r0.y, r0.z, r0.w, r1.x, r1.y, r1.z, r1.w,
                    r2.x, r2.y, r2.z, r2.w, r3.x, r3.y, r3.z, r3.w};
    float* fd = flow + ((size_t)b * NN + d) * LD_F;
#pragma unroll
    for (int q = 0; q < QQ; ++q) atomicAdd(&fd[q], a * fv[q]);
}

// stream+collide update, moments, predictions, var^2 accumulation (LDS reduce)
__global__ void update_kernel(float* __restrict__ f, const float* __restrict__ flow,
                              const float* __restrict__ omega,
                              const float* __restrict__ xi,
                              float* __restrict__ out, float* __restrict__ vsum, int t) {
    __shared__ float red[256];
    int idx = blockIdx.x * blockDim.x + threadIdx.x;  // over B*N
    float v2 = 0.0f;
    if (idx < BN) {
        int b = idx / NN, n = idx - b * NN;
        size_t base = (size_t)idx * LD_F;
        const float4* fp = (const float4*)(f + base);
        const float4* lp = (const float4*)(flow + base);
        const float4* op = (const float4*)(omega + base);
        float dens = 0.0f, m1 = 0.0f, m2 = 0.0f;
        float4 fo4[4];
#pragma unroll
        for (int i = 0; i < 4; ++i) {
            float4 fo = fp[i], fl = lp[i], om = op[i];
            float* fe = (float*)&fo;
            const float* le = (const float*)&fl;
            const float* oe = (const float*)&om;
#pragma unroll
            for (int j = 0; j < 4; ++j) {
                int q = i * 4 + j;
                float fn = fe[j] + DT * (le[j] - fe[j] + oe[j]);
                fn = fmaxf(fn, 0.0f);
                fe[j] = fn;                         // q==15 stays 0
                float x = (q < QQ) ? xi[q] : 0.0f;
                dens += fn;
                m1   += fn * x;
                m2   += fn * x * x;
            }
            fo4[i] = fo;
        }
        float4* fw = (float4*)(f + base);
#pragma unroll
        for (int i = 0; i < 4; ++i) fw[i] = fo4[i];
        float inv = 1.0f / fmaxf(dens, EPSF);
        float vel = m1 * inv;
        float var = m2 * inv - vel * vel;
        size_t ob = ((size_t)(b * OUT_LEN + t) * NN + n) * 2;
        out[ob + 0] = dens;
        out[ob + 1] = vel;
        v2 = var * var;
    }
    red[threadIdx.x] = v2;
    __syncthreads();
#pragma unroll
    for (int s = 128; s > 0; s >>= 1) {
        if ((int)threadIdx.x < s) red[threadIdx.x] += red[threadIdx.x + s];
        __syncthreads();
    }
    if (threadIdx.x == 0) atomicAdd(vsum, red[0]);
}

__global__ void finalize_kernel(const float* __restrict__ vsum, float* __restrict__ out) {
    if (threadIdx.x == 0 && blockIdx.x == 0)
        out[(size_t)BB * OUT_LEN * NN * 2] =
            vsum[0] * (1.0f / ((float)OUT_LEN * (float)BB * (float)NN));
}

// ---------------------------------------------------------------------------
extern "C" void kernel_launch(void* const* d_in, const int* in_sizes, int n_in,
                              void* d_out, int out_size, void* d_ws, size_t ws_size,
                              hipStream_t stream) {
    const float* inputs = (const float*)d_in[0];
    const int*   src    = (const int*)  d_in[1];
    const int*   dst    = (const int*)  d_in[2];
    const float* edge_w = (const float*)d_in[3];
    const float* W1     = (const float*)d_in[4];
    const float* b1v    = (const float*)d_in[5];
    const float* W2     = (const float*)d_in[6];
    const float* b2v    = (const float*)d_in[7];
    const float* Wc1    = (const float*)d_in[8];
    const float* bc1    = (const float*)d_in[9];
    const float* Wc2    = (const float*)d_in[10];
    const float* bc2    = (const float*)d_in[11];
    const float* xi     = (const float*)d_in[12];
    float* out = (float*)d_out;

    // workspace carving (float offsets; all region starts multiple of 8 floats)
    float* ws     = (float*)d_ws;
    float* deg    = ws;                      // 10000
    float* wsum   = ws + 10000;              // 10000
    float* norm   = ws + 20000;              // 10000
    float* vsum   = ws + 30000;              // 1 (padded to 30032)
    float* alpha  = ws + 30032;              // 160000              -> 190032
    float* cat1   = ws + 190032;             // BN*8   = 320000     -> 510032
    float* f      = ws + 510032;             // BN*16  = 640000     -> 1150032
    float* omega  = ws + 1150032;            // BN*16  = 640000     -> 1790032
    float* flow   = ws + 1790032;            // BN*16  = 640000     -> 2430032
    float* hidden = ws + 2430032;            // BN*128 = 5120000    -> 7550032
    _Float16* W1p  = (_Float16*)(ws + 7550032);  // 2048 f
    _Float16* Wc1p = (_Float16*)(ws + 7552080);  // 2048 f
    _Float16* W2p  = (_Float16*)(ws + 7554128);  // 3072 f
    _Float16* Wc2p = (_Float16*)(ws + 7557200);  // 1024 f
    float* cat2   = ws + 7558224;            // BN*384 = 15360000   -> 22918224

    const int TB = 256;

    // zero accumulators (deg, wsum, vsum) and hop-target region of cat2
    hipMemsetAsync(ws, 0, 30040 * sizeof(float), stream);
    hipMemsetAsync(cat2, 0, (size_t)BN * LD_C2 * sizeof(float), stream);

    // pack weight matrices into f16 fragment layout
    pack_b<<<1, 256, 0, stream>>>(W1, HH, 6, 1, 8, W1p);     // [6,128]
    pack_b<<<1, 256, 0, stream>>>(Wc1, HH, QQ, 1, 8, Wc1p);  // [15,128]
    pack_b<<<2, 256, 0, stream>>>(W2, QQ, 384, 12, 1, W2p);  // [384,15]
    pack_b<<<1, 256, 0, stream>>>(Wc2, QQ, HH, 4, 1, Wc2p);  // [128,15]

    // graph stats
    deg_wsum_kernel<<<(EE + TB - 1) / TB, TB, 0, stream>>>(dst, edge_w, deg, wsum);
    norm_kernel<<<(NN + TB - 1) / TB, TB, 0, stream>>>(deg, norm);
    alpha_kernel<<<(EE + TB - 1) / TB, TB, 0, stream>>>(edge_w, dst, wsum, alpha);

    // ---- encoder: diff_conv #1 on [N,2] features -------------------------
    extract_kernel<<<(BN + TB - 1) / TB, TB, 0, stream>>>(inputs, cat1);
    {
        dim3 gs((EE * 2 + TB - 1) / TB, BB);
        dim3 gn((NN * 2 + TB - 1) / TB, BB);
        hop_scatter<2, LD_C1><<<gs, TB, 0, stream>>>(cat1, 0, 2, norm, src, dst);
        hop_scale  <2, LD_C1><<<gn, TB, 0, stream>>>(cat1, 2, norm);
        hop_scatter<2, LD_C1><<<gs, TB, 0, stream>>>(cat1, 2, 4, norm, src, dst);
        hop_scale  <2, LD_C1><<<gn, TB, 0, stream>>>(cat1, 4, norm);
    }
    // x = relu(cat1 @ W1 + b1) -> cat2[:, 0:128]
    gemm_wide<1, 8, 2, LD_C1, LD_C2><<<MTILES, TB, 0, stream>>>(
        cat1, (const v16h*)W1p, b1v, cat2);

    // ---- encoder: diff_conv #2 on [N,128] features -----------------------
    {
        dim3 gs((EE * 128 + TB - 1) / TB, BB);
        dim3 gn((NN * 128 + TB - 1) / TB, BB);
        hop_scatter<128, LD_C2><<<gs, TB, 0, stream>>>(cat2, 0, 128, norm, src, dst);
        hop_scale  <128, LD_C2><<<gn, TB, 0, stream>>>(cat2, 128, norm);
        hop_scatter<128, LD_C2><<<gs, TB, 0, stream>>>(cat2, 128, 256, norm, src, dst);
        hop_scale  <128, LD_C2><<<gn, TB, 0, stream>>>(cat2, 256, norm);
    }
    // f0 = softplus(cat2 @ W2 + b2)  (12 unrolled WMMAs)
    gemm_narrow<3, 384, LD_C2, LD_F><<<(MTILES + 7) / 8, TB, 0, stream>>>(
        cat2, (const v16h*)W2p, b2v, QQ, f);

    // ---- 12 lattice-Boltzmann steps --------------------------------------
    for (int t = 0; t < OUT_LEN; ++t) {
        // hidden = tanh(f @ Wc1 + bc1)
        gemm_wide<2, 16, 1, LD_F, HH><<<MTILES, TB, 0, stream>>>(
            f, (const v16h*)Wc1p, bc1, hidden);
        // omega = hidden @ Wc2 + bc2  (4 unrolled WMMAs)
        gemm_narrow<0, 128, HH, LD_F><<<(MTILES + 7) / 8, TB, 0, stream>>>(
            hidden, (const v16h*)Wc2p, bc2, QQ, omega);
        // flow_in = segment_sum(alpha * f[src] -> dst)
        hipMemsetAsync(flow, 0, (size_t)BN * LD_F * sizeof(float), stream);
        {
            dim3 g((EE + TB - 1) / TB, BB);
            flow_scatter<<<g, TB, 0, stream>>>(f, alpha, src, dst, flow);
        }
        // stream + collide, moments, predictions, var^2
        update_kernel<<<(BN + TB - 1) / TB, TB, 0, stream>>>(f, flow, omega, xi, out,
                                                             vsum, t);
    }
    finalize_kernel<<<1, 32, 0, stream>>>(vsum, out);
}
